// GNNEmbedder_1013612281974
// MI455X (gfx1250) — matrix operand order repaired and checked by reading, not compiled, not run on hardware
//
#include <hip/hip_runtime.h>
#include <hip/hip_bf16.h>

#define N_NODES 50000
#define N_EDGES 800000
#define EN_TOT  (N_EDGES + N_NODES)   // edges + self loops
#define IN_DIMM 128
#define DIMM    64
#define HEADS   4
#define NEG_SLOPE 0.2f

typedef __attribute__((ext_vector_type(16))) __bf16 v16bf;
typedef __attribute__((ext_vector_type(8)))  float  v8f;

union BfVec { v16bf v; unsigned short u[16]; uint4 q[2]; };

__device__ __forceinline__ unsigned short f2bf(float f) {
    unsigned u = __float_as_uint(f);
    u += 0x7FFFu + ((u >> 16) & 1u);          // round-to-nearest-even
    return (unsigned short)(u >> 16);
}
__device__ __forceinline__ float bf2f(unsigned short h) {
    return __uint_as_float(((unsigned)h) << 16);
}

// ---------------------------------------------------------------- weight prep
// Convert f32 weight [K, Nout] into hi/lo bf16 pre-swizzled into the WMMA
// B-fragment register layout: element t = (((ct*(K/32)+kc)*32 + lane)*16 + i)
// holds W[kc*32 + (lane>>4)*16 + i][ct*16 + (lane&15)].
__global__ void k_prep_w(const float* __restrict__ W, unsigned short* __restrict__ Wh,
                         unsigned short* __restrict__ Wl, int K, int Nout)
{
    int t = blockIdx.x * blockDim.x + threadIdx.x;
    if (t >= K * Nout) return;
    int chunks = K >> 5;
    int i    = t & 15;
    int lane = (t >> 4) & 31;
    int rest = t >> 9;
    int kc = rest % chunks;
    int ct = rest / chunks;
    int n = ct * 16 + (lane & 15);
    int k = kc * 32 + (lane >> 4) * 16 + i;
    float f = W[k * Nout + n];
    unsigned short h = f2bf(f);
    Wh[t] = h;
    Wl[t] = f2bf(f - bf2f(h));
}

__global__ void k_zero(float* __restrict__ p, long long n) {
    long long i = (long long)blockIdx.x * blockDim.x + threadIdx.x;
    if (i < n) p[i] = 0.0f;
}

// ---------------------------------------------------------------- degrees
__global__ void k_init_deg(float* __restrict__ deg) {
    int i = blockIdx.x * blockDim.x + threadIdx.x;
    if (i < N_NODES) deg[i] = 1.0f;           // self loop
}
__global__ void k_count_deg(const int* __restrict__ dst, float* __restrict__ deg) {
    int e = blockIdx.x * blockDim.x + threadIdx.x;
    if (e < N_EDGES) atomicAdd(&deg[dst[e]], 1.0f);
}
__global__ void k_deg_to_dinv(float* __restrict__ deg) {
    int i = blockIdx.x * blockDim.x + threadIdx.x;
    if (i < N_NODES) deg[i] = rsqrtf(deg[i]); // deg >= 1 always
}

// ---------------------------------------------------------------- WMMA GEMM
// out[M=50000, Nout] = A[M,K] * W[K,Nout] (+bias)(+relu), bf16x3 split:
// D = Ah*Bh + Ah*Bl + Al*Bh  (~fp32 accuracy, f32 accumulate)
// one wave32 per 16x16 tile; grid.x = 3125 row tiles, grid.y = Nout/16
__global__ void __launch_bounds__(32)
k_gemm_bf16x3(const float* __restrict__ A,
              const unsigned short* __restrict__ Wh, const unsigned short* __restrict__ Wl,
              const float* __restrict__ bias, float* __restrict__ out,
              int K, int Nout, int relu)
{
    const int lane   = threadIdx.x;          // 0..31
    const int half   = lane >> 4;            // 0 or 1
    const int l15    = lane & 15;
    const int mbase  = blockIdx.x * 16;
    const int ncol   = blockIdx.y * 16 + l15;
    const int chunks = K >> 5;

    const float* arow = A + (long long)(mbase + l15) * K;

    v8f acc = {};
    for (int kc = 0; kc < chunks; ++kc) {
        // A fragment = one contiguous 16-float run: [kb+half*16, +16)
        const float4* p = (const float4*)(arow + (kc << 5) + (half << 4));
        float fa[16];
        *(float4*)(fa + 0)  = p[0];
        *(float4*)(fa + 4)  = p[1];
        *(float4*)(fa + 8)  = p[2];
        *(float4*)(fa + 12) = p[3];
        BfVec ah, al;
        #pragma unroll
        for (int i = 0; i < 16; ++i) {
            unsigned short h = f2bf(fa[i]);
            ah.u[i] = h;
            al.u[i] = f2bf(fa[i] - bf2f(h));
        }
        // B fragment: pre-swizzled, 32B contiguous per lane -> 2x b128 each
        long long bb = ((long long)(blockIdx.y * chunks + kc) * 32 + lane) * 16;
        BfVec bh, bl;
        bh.q[0] = ((const uint4*)(Wh + bb))[0];
        bh.q[1] = ((const uint4*)(Wh + bb))[1];
        bl.q[0] = ((const uint4*)(Wl + bb))[0];
        bl.q[1] = ((const uint4*)(Wl + bb))[1];

        acc = __builtin_amdgcn_wmma_f32_16x16x32_bf16(false, ah.v, false, bh.v, (short)0, acc, false, false);
        acc = __builtin_amdgcn_wmma_f32_16x16x32_bf16(false, ah.v, false, bl.v, (short)0, acc, false, false);
        acc = __builtin_amdgcn_wmma_f32_16x16x32_bf16(false, al.v, false, bh.v, (short)0, acc, false, false);
    }

    const float bv = bias ? bias[ncol] : 0.0f;
    #pragma unroll
    for (int r = 0; r < 8; ++r) {           // C/D: VGPR r -> M = r + half*8
        int m = mbase + (half << 3) + r;
        float v = acc[r] + bv;
        if (relu) v = fmaxf(v, 0.0f);
        out[(long long)m * Nout + ncol] = v;
    }
}

// ---------------------------------------------------------------- GCN scatter
__global__ void k_gcn_edge(const int* __restrict__ src, const int* __restrict__ dst,
                           const float* __restrict__ dinv, const float* __restrict__ hw,
                           float* __restrict__ out)
{
    long long t = (long long)blockIdx.x * blockDim.x + threadIdx.x;
    if (t >= (long long)N_EDGES * DIMM) return;
    int e = (int)(t >> 6), c = (int)(t & 63);
    int s = src[e], d = dst[e];
    float norm = dinv[s] * dinv[d];
    atomicAdd(&out[(long long)d * DIMM + c], hw[(long long)s * DIMM + c] * norm);
}

// accum + self-loop term + bias, relu -> h  (in place on accum is safe)
__global__ void k_gcn_fin(const float* __restrict__ accum, const float* __restrict__ hw,
                          const float* __restrict__ dinv, const float* __restrict__ b,
                          float* __restrict__ hout)
{
    long long t = (long long)blockIdx.x * blockDim.x + threadIdx.x;
    if (t >= (long long)N_NODES * DIMM) return;
    int i = (int)(t >> 6), c = (int)(t & 63);
    float di = dinv[i];
    float v = accum[t] + hw[t] * di * di + b[c];
    hout[t] = fmaxf(v, 0.0f);
}

// ---------------------------------------------------------------- GAT
// per (node, head) attention coefficients: one wave per pair, shuffle reduce
__global__ void k_att_coeff(const float* __restrict__ G, const float* __restrict__ att_src,
                            const float* __restrict__ att_dst,
                            float* __restrict__ asrc, float* __restrict__ adst)
{
    int wid  = blockIdx.x * (blockDim.x >> 5) + (threadIdx.x >> 5);
    int lane = threadIdx.x & 31;
    if (wid >= N_NODES * HEADS) return;
    int n = wid >> 2, h = wid & 3;
    const float* g  = G + (long long)n * (HEADS * DIMM) + h * DIMM;
    const float* as = att_src + h * DIMM;
    const float* ad = att_dst + h * DIMM;
    float g0 = g[lane], g1 = g[lane + 32];
    float s1 = g0 * as[lane] + g1 * as[lane + 32];
    float s2 = g0 * ad[lane] + g1 * ad[lane + 32];
    #pragma unroll
    for (int off = 16; off > 0; off >>= 1) {
        s1 += __shfl_down(s1, off, 32);
        s2 += __shfl_down(s2, off, 32);
    }
    if (lane == 0) { asrc[wid] = s1; adst[wid] = s2; }
}

__global__ void k_gat_prep(unsigned* __restrict__ maxkey, float* __restrict__ denom) {
    int i = blockIdx.x * blockDim.x + threadIdx.x;
    if (i < N_NODES * HEADS) { maxkey[i] = 0u; denom[i] = 0.0f; }
}

__device__ __forceinline__ unsigned fkey(float f) {
    unsigned u = __float_as_uint(f);
    return (u & 0x80000000u) ? ~u : (u | 0x80000000u);   // monotone float->uint
}
__device__ __forceinline__ float fkey_inv(unsigned k) {
    unsigned u = (k & 0x80000000u) ? (k & 0x7FFFFFFFu) : ~k;
    return __uint_as_float(u);
}

__global__ void k_gat_logit(const int* __restrict__ src, const int* __restrict__ dst,
                            const float* __restrict__ asrc, const float* __restrict__ adst,
                            float* __restrict__ elog, unsigned* __restrict__ maxkey)
{
    int e = blockIdx.x * blockDim.x + threadIdx.x;
    if (e >= EN_TOT) return;
    int s = (e < N_EDGES) ? src[e] : (e - N_EDGES);
    int d = (e < N_EDGES) ? dst[e] : (e - N_EDGES);
    #pragma unroll
    for (int h = 0; h < HEADS; ++h) {
        float v = asrc[s * HEADS + h] + adst[d * HEADS + h];
        v = (v > 0.0f) ? v : NEG_SLOPE * v;               // leaky relu
        elog[(long long)e * HEADS + h] = v;
        atomicMax(&maxkey[d * HEADS + h], fkey(v));
    }
}

__global__ void k_gat_decode_max(const unsigned* __restrict__ maxkey, float* __restrict__ maxv) {
    int i = blockIdx.x * blockDim.x + threadIdx.x;
    if (i < N_NODES * HEADS) maxv[i] = fkey_inv(maxkey[i]);
}

__global__ void k_gat_exp(const int* __restrict__ dst, const float* __restrict__ maxv,
                          float* __restrict__ elog, float* __restrict__ denom)
{
    int e = blockIdx.x * blockDim.x + threadIdx.x;
    if (e >= EN_TOT) return;
    int d = (e < N_EDGES) ? dst[e] : (e - N_EDGES);
    #pragma unroll
    for (int h = 0; h < HEADS; ++h) {
        float ex = expf(elog[(long long)e * HEADS + h] - maxv[d * HEADS + h]);
        elog[(long long)e * HEADS + h] = ex;
        atomicAdd(&denom[d * HEADS + h], ex);
    }
}

// weighted aggregation; 4 heads folded into a single atomic per (edge, channel)
__global__ void k_gat_agg(const int* __restrict__ src, const int* __restrict__ dst,
                          const float* __restrict__ G, const float* __restrict__ elog,
                          const float* __restrict__ denom, float* __restrict__ out)
{
    long long t = (long long)blockIdx.x * blockDim.x + threadIdx.x;
    if (t >= (long long)EN_TOT * DIMM) return;
    int e = (int)(t >> 6), c = (int)(t & 63);
    int s = (e < N_EDGES) ? src[e] : (e - N_EDGES);
    int d = (e < N_EDGES) ? dst[e] : (e - N_EDGES);
    const float4 ex  = *(const float4*)(elog  + (long long)e * HEADS);
    const float4 den = *(const float4*)(denom + (long long)d * HEADS);
    const float* gs = G + (long long)s * (HEADS * DIMM) + c;
    float sum = gs[0 * DIMM] * (ex.x / fmaxf(den.x, 1e-16f))
              + gs[1 * DIMM] * (ex.y / fmaxf(den.y, 1e-16f))
              + gs[2 * DIMM] * (ex.z / fmaxf(den.z, 1e-16f))
              + gs[3 * DIMM] * (ex.w / fmaxf(den.w, 1e-16f));
    atomicAdd(&out[(long long)d * DIMM + c], 0.25f * sum);   // mean over heads
}

__global__ void k_add_bias(const float* __restrict__ in, const float* __restrict__ b,
                           float* __restrict__ out)
{
    long long t = (long long)blockIdx.x * blockDim.x + threadIdx.x;
    if (t >= (long long)N_NODES * DIMM) return;
    out[t] = in[t] + b[(int)(t & 63)];
}

// ---------------------------------------------------------------- launch
static inline int cdiv_i(long long a, int b) { return (int)((a + b - 1) / b); }

extern "C" void kernel_launch(void* const* d_in, const int* in_sizes, int n_in,
                              void* d_out, int out_size, void* d_ws, size_t ws_size,
                              hipStream_t stream) {
    (void)in_sizes; (void)n_in; (void)out_size; (void)ws_size;

    const float* x       = (const float*)d_in[0];
    const int*   ei      = (const int*)d_in[1];
    const float* w_in    = (const float*)d_in[2];
    const float* b_in    = (const float*)d_in[3];
    const float* w_g1    = (const float*)d_in[4];
    const float* b_g1    = (const float*)d_in[5];
    const float* w_g2    = (const float*)d_in[6];
    const float* b_g2    = (const float*)d_in[7];
    const float* w_gat   = (const float*)d_in[8];
    const float* att_src = (const float*)d_in[9];
    const float* att_dst = (const float*)d_in[10];
    const float* b_gat   = (const float*)d_in[11];
    const float* w_out   = (const float*)d_in[12];
    const float* b_out   = (const float*)d_in[13];
    float* out = (float*)d_out;

    const int* src = ei;
    const int* dst = ei + N_EDGES;

    // workspace carve-out
    char* wp = (char*)d_ws;
    auto alloc = [&](size_t bytes) -> void* {
        void* p = (void*)wp;
        wp += (bytes + 255) & ~(size_t)255;
        return p;
    };
    unsigned short* wh_in  = (unsigned short*)alloc((size_t)IN_DIMM * DIMM * 2);
    unsigned short* wl_in  = (unsigned short*)alloc((size_t)IN_DIMM * DIMM * 2);
    unsigned short* wh_g1  = (unsigned short*)alloc((size_t)DIMM * DIMM * 2);
    unsigned short* wl_g1  = (unsigned short*)alloc((size_t)DIMM * DIMM * 2);
    unsigned short* wh_g2  = (unsigned short*)alloc((size_t)DIMM * DIMM * 2);
    unsigned short* wl_g2  = (unsigned short*)alloc((size_t)DIMM * DIMM * 2);
    unsigned short* wh_gat = (unsigned short*)alloc((size_t)DIMM * HEADS * DIMM * 2);
    unsigned short* wl_gat = (unsigned short*)alloc((size_t)DIMM * HEADS * DIMM * 2);
    unsigned short* wh_out = (unsigned short*)alloc((size_t)DIMM * DIMM * 2);
    unsigned short* wl_out = (unsigned short*)alloc((size_t)DIMM * DIMM * 2);
    float* dinv   = (float*)alloc((size_t)N_NODES * 4);
    float* T0     = (float*)alloc((size_t)N_NODES * DIMM * 4);
    float* T1     = (float*)alloc((size_t)N_NODES * DIMM * 4);
    float* T2     = (float*)alloc((size_t)N_NODES * DIMM * 4);
    float* G      = (float*)alloc((size_t)N_NODES * HEADS * DIMM * 4);
    float* elog   = (float*)alloc((size_t)EN_TOT * HEADS * 4);
    unsigned* mkey= (unsigned*)alloc((size_t)N_NODES * HEADS * 4);
    float* maxv   = (float*)alloc((size_t)N_NODES * HEADS * 4);
    float* denom  = (float*)alloc((size_t)N_NODES * HEADS * 4);
    float* asrc   = (float*)alloc((size_t)N_NODES * HEADS * 4);
    float* adst   = (float*)alloc((size_t)N_NODES * HEADS * 4);

    const int B = 256;
    const int rowTiles = N_NODES / 16;                 // 3125 exactly
    const long long ND = (long long)N_NODES * DIMM;    // 3.2M
    const long long ED = (long long)N_EDGES * DIMM;    // 51.2M
    const long long AD = (long long)EN_TOT * DIMM;     // 54.4M

    // weights -> hi/lo bf16, pre-swizzled into WMMA B-fragment layout
    k_prep_w<<<cdiv_i(IN_DIMM * DIMM, B), B, 0, stream>>>(w_in,  wh_in,  wl_in,  IN_DIMM, DIMM);
    k_prep_w<<<cdiv_i(DIMM * DIMM, B), B, 0, stream>>>(w_g1,  wh_g1,  wl_g1,  DIMM, DIMM);
    k_prep_w<<<cdiv_i(DIMM * DIMM, B), B, 0, stream>>>(w_g2,  wh_g2,  wl_g2,  DIMM, DIMM);
    k_prep_w<<<cdiv_i(DIMM * HEADS * DIMM, B), B, 0, stream>>>(w_gat, wh_gat, wl_gat, DIMM, HEADS * DIMM);
    k_prep_w<<<cdiv_i(DIMM * DIMM, B), B, 0, stream>>>(w_out, wh_out, wl_out, DIMM, DIMM);

    // symmetric-normalization degrees
    k_init_deg<<<cdiv_i(N_NODES, B), B, 0, stream>>>(dinv);
    k_count_deg<<<cdiv_i(N_EDGES, B), B, 0, stream>>>(dst, dinv);
    k_deg_to_dinv<<<cdiv_i(N_NODES, B), B, 0, stream>>>(dinv);

    // h0 = relu(x @ w_in + b_in)
    k_gemm_bf16x3<<<dim3(rowTiles, DIMM / 16), 32, 0, stream>>>(x, wh_in, wl_in, b_in, T0, IN_DIMM, DIMM, 1);

    // GCN layer 1: h1 = relu(scatter(h0 @ w_g1) + b_g1)
    k_gemm_bf16x3<<<dim3(rowTiles, DIMM / 16), 32, 0, stream>>>(T0, wh_g1, wl_g1, nullptr, T1, DIMM, DIMM, 0);
    k_zero<<<cdiv_i(ND, B), B, 0, stream>>>(T2, ND);
    k_gcn_edge<<<cdiv_i(ED, B), B, 0, stream>>>(src, dst, dinv, T1, T2);
    k_gcn_fin<<<cdiv_i(ND, B), B, 0, stream>>>(T2, T1, dinv, b_g1, T2);

    // GCN layer 2: h2 = relu(scatter(h1 @ w_g2) + b_g2)
    k_gemm_bf16x3<<<dim3(rowTiles, DIMM / 16), 32, 0, stream>>>(T2, wh_g2, wl_g2, nullptr, T1, DIMM, DIMM, 0);
    k_zero<<<cdiv_i(ND, B), B, 0, stream>>>(T0, ND);
    k_gcn_edge<<<cdiv_i(ED, B), B, 0, stream>>>(src, dst, dinv, T1, T0);
    k_gcn_fin<<<cdiv_i(ND, B), B, 0, stream>>>(T0, T1, dinv, b_g2, T0);

    // GAT: G = h2 @ w_gat  [N, 4*64]
    k_gemm_bf16x3<<<dim3(rowTiles, HEADS * DIMM / 16), 32, 0, stream>>>(T0, wh_gat, wl_gat, nullptr, G, DIMM, HEADS * DIMM, 0);
    k_att_coeff<<<cdiv_i((long long)N_NODES * HEADS, B / 32), B, 0, stream>>>(G, att_src, att_dst, asrc, adst);
    k_gat_prep<<<cdiv_i(N_NODES * HEADS, B), B, 0, stream>>>(mkey, denom);
    k_gat_logit<<<cdiv_i(EN_TOT, B), B, 0, stream>>>(src, dst, asrc, adst, elog, mkey);
    k_gat_decode_max<<<cdiv_i(N_NODES * HEADS, B), B, 0, stream>>>(mkey, maxv);
    k_gat_exp<<<cdiv_i(EN_TOT, B), B, 0, stream>>>(dst, maxv, elog, denom);
    k_zero<<<cdiv_i(ND, B), B, 0, stream>>>(T2, ND);
    k_gat_agg<<<cdiv_i(AD, B), B, 0, stream>>>(src, dst, G, elog, denom, T2);
    k_add_bias<<<cdiv_i(ND, B), B, 0, stream>>>(T2, b_gat, T1);

    // output projection -> d_out
    k_gemm_bf16x3<<<dim3(rowTiles, DIMM / 16), 32, 0, stream>>>(T1, wh_out, wl_out, b_out, out, DIMM, DIMM, 0);
}
